// LSTM_18975165514426
// MI455X (gfx1250) — matrix operand order repaired and checked by reading, not compiled, not run on hardware
//
#include <hip/hip_runtime.h>
#include <stdint.h>

// ---------------------------------------------------------------------------
// LSTM on MI455X (gfx1250): bf16 WMMA (16x16x32) recurrent GEMMs, f32 state.
//   T=512 sequential steps; per step 4 gate tiles per workgroup (1 wave/gate),
//   fused pointwise c/h update through LDS. No software prefetch: the
//   compiler's load pipeline (5-7 b128 loads in flight ahead of each WMMA)
//   covers L2 latency, and GL2-scope prefetches are redundant for the
//   L2-resident 12.6 MB weight set.
// ---------------------------------------------------------------------------

typedef __bf16 v8bf  __attribute__((ext_vector_type(8)));
typedef __bf16 v16bf __attribute__((ext_vector_type(16)));
typedef float  v8f   __attribute__((ext_vector_type(8)));

#define T_ 512
#define B_ 64
#define E_ 512
#define H_ 1024
#define K_ (E_ + H_)  // 1536

__device__ __forceinline__ uint16_t f2bf(float f) {
  // round-to-nearest-even f32 -> bf16 (bits)
  uint32_t u = __float_as_uint(f);
  u += 0x7FFFu + ((u >> 16) & 1u);
  return (uint16_t)(u >> 16);
}

__device__ __forceinline__ float sigm(float x) {
  return 1.0f / (1.0f + __expf(-x));
}

// ---- one-time converts -----------------------------------------------------

__global__ void cvt_weights(const float* __restrict__ wg,
                            const float* __restrict__ wi,
                            const float* __restrict__ wf,
                            const float* __restrict__ wo,
                            uint16_t* __restrict__ Wb) {
  size_t idx = (size_t)blockIdx.x * 256 + threadIdx.x;
  const size_t per = (size_t)H_ * K_;
  if (idx >= 4 * per) return;
  int gate = (int)(idx / per);
  size_t off = idx % per;
  const float* src = (gate == 0) ? wg : (gate == 1) ? wi : (gate == 2) ? wf : wo;
  Wb[idx] = f2bf(src[off]);
}

__global__ void cvt_embeds(const float* __restrict__ e,
                           uint16_t* __restrict__ Xb,
                           uint16_t* __restrict__ Hb0) {
  size_t idx = (size_t)blockIdx.x * 256 + threadIdx.x;
  const size_t n = (size_t)T_ * B_ * E_;
  if (idx < n) Xb[idx] = f2bf(e[idx]);
  if (idx < (size_t)B_ * H_) Hb0[idx] = 0;  // h0 = 0 (bf16 zero bits)
}

// ---- one recurrent step ----------------------------------------------------
// grid: (H_/16, B_/16) = (64, 4); block: 128 threads = 4 waves, wave == gate.

__global__ __launch_bounds__(128) void lstm_step(
    const uint16_t* __restrict__ Xb,   // [T][B][E] bf16
    const uint16_t* __restrict__ Hin,  // [B][H]   bf16 (h_{t-1})
    uint16_t* __restrict__ Hout,       // [B][H]   bf16 (h_t)
    const uint16_t* __restrict__ Wb,   // [4][H][K] bf16
    const float* __restrict__ bg, const float* __restrict__ bi,
    const float* __restrict__ bfv, const float* __restrict__ bo,
    float* __restrict__ h_seq,         // [T][B][H] f32
    float* __restrict__ c_seq,         // [T][B][H] f32
    int t) {
  __shared__ float lds[4][16][16];

  const int tid  = threadIdx.x;
  const int gate = tid >> 5;       // wave id == gate (g,i,f,o)
  const int lane = tid & 31;
  const int half = lane >> 4;      // 0: lanes 0-15, 1: lanes 16-31
  const int l15  = lane & 15;
  const int n_base = blockIdx.x * 16;  // column tile within H
  const int m_base = blockIdx.y * 16;  // row (batch) tile within B

  // A operand row for this lane (batch row), B operand row (weight row n)
  const uint16_t* __restrict__ xrow = Xb + ((size_t)t * B_ + (m_base + l15)) * E_;
  const uint16_t* __restrict__ hrow = Hin + (size_t)(m_base + l15) * H_;
  const uint16_t* __restrict__ wrow = Wb + ((size_t)gate * H_ + (n_base + l15)) * K_;

  v8f acc = {};

  // K region 1: x part (k in [0, E_)); E_ is a multiple of 32 so no straddle.
  for (int k0 = 0; k0 < E_; k0 += 32) {
    const int klo = k0 + half * 16;
    v8bf alo = *reinterpret_cast<const v8bf*>(xrow + klo);
    v8bf ahi = *reinterpret_cast<const v8bf*>(xrow + klo + 8);
    v16bf a = __builtin_shufflevector(alo, ahi, 0, 1, 2, 3, 4, 5, 6, 7,
                                      8, 9, 10, 11, 12, 13, 14, 15);
    v16bf b = *reinterpret_cast<const v16bf*>(wrow + k0 + half * 16);
    acc = __builtin_amdgcn_wmma_f32_16x16x32_bf16(false, a, false, b,
                                                  (short)0, acc, false, false);
  }
  // K region 2: h part (k in [E_, K_))
  for (int k0 = E_; k0 < K_; k0 += 32) {
    const int klo = (k0 - E_) + half * 16;
    v8bf alo = *reinterpret_cast<const v8bf*>(hrow + klo);
    v8bf ahi = *reinterpret_cast<const v8bf*>(hrow + klo + 8);
    v16bf a = __builtin_shufflevector(alo, ahi, 0, 1, 2, 3, 4, 5, 6, 7,
                                      8, 9, 10, 11, 12, 13, 14, 15);
    v16bf b = *reinterpret_cast<const v16bf*>(wrow + k0 + half * 16);
    acc = __builtin_amdgcn_wmma_f32_16x16x32_bf16(false, a, false, b,
                                                  (short)0, acc, false, false);
  }

  // C/D layout: VGPR r, lane l -> (m = r + 8*half, n = l15)
#pragma unroll
  for (int r = 0; r < 8; ++r) lds[gate][r + half * 8][l15] = acc[r];
  __syncthreads();

  // Fused pointwise LSTM cell update: 256 elements over 128 threads.
  for (int e = tid; e < 256; e += 128) {
    const int m = e >> 4, n = e & 15;
    const int b = m_base + m;
    const int hh = n_base + n;
    const float g = tanhf(lds[0][m][n] + bg[hh]);
    const float i = sigm(lds[1][m][n] + bi[hh]);
    const float f = sigm(lds[2][m][n] + bfv[hh]);
    const float o = sigm(lds[3][m][n] + bo[hh]);
    const size_t cur = ((size_t)t * B_ + b) * (size_t)H_ + hh;
    const float cp = (t == 0) ? 0.0f : c_seq[cur - (size_t)B_ * H_];
    const float c = f * cp + i * g;
    const float h = o * tanhf(c);
    h_seq[cur] = h;
    c_seq[cur] = c;
    Hout[(size_t)b * H_ + hh] = f2bf(h);
  }
}

// ---------------------------------------------------------------------------

extern "C" void kernel_launch(void* const* d_in, const int* in_sizes, int n_in,
                              void* d_out, int out_size, void* d_ws,
                              size_t ws_size, hipStream_t stream) {
  (void)in_sizes; (void)n_in; (void)out_size; (void)ws_size;

  const float* embeds = (const float*)d_in[0];
  const float* Wg = (const float*)d_in[1];
  const float* Wi = (const float*)d_in[2];
  const float* Wf = (const float*)d_in[3];
  const float* Wo = (const float*)d_in[4];
  const float* bg = (const float*)d_in[5];
  const float* bi = (const float*)d_in[6];
  const float* bf = (const float*)d_in[7];
  const float* bo = (const float*)d_in[8];

  float* h_seq = (float*)d_out;
  float* c_seq = h_seq + (size_t)T_ * B_ * H_;

  // Workspace carve-up (bf16 buffers): weights | embeds | h ping-pong
  uint16_t* Wb = (uint16_t*)d_ws;                 // 4*H*K      = 12.58 MB
  uint16_t* Xb = Wb + (size_t)4 * H_ * K_;        // T*B*E      = 33.55 MB
  uint16_t* Hb = Xb + (size_t)T_ * B_ * E_;       // 2*B*H      =  0.26 MB

  {
    size_t n = (size_t)4 * H_ * K_;
    cvt_weights<<<(unsigned)((n + 255) / 256), 256, 0, stream>>>(Wg, Wi, Wf, Wo, Wb);
  }
  {
    size_t n = (size_t)T_ * B_ * E_;
    cvt_embeds<<<(unsigned)((n + 255) / 256), 256, 0, stream>>>(embeds, Xb, Hb);
  }

  dim3 grid(H_ / 16, B_ / 16);  // 64 x 4 tiles
  for (int t = 0; t < T_; ++t) {
    uint16_t* hin  = Hb + (size_t)(t & 1) * B_ * H_;
    uint16_t* hout = Hb + (size_t)((t & 1) ^ 1) * B_ * H_;
    lstm_step<<<grid, 128, 0, stream>>>(Xb, hin, hout, Wb, bg, bi, bf, bo,
                                        h_seq, c_seq, t);
  }
}
